// SelfAttention_1898375545413
// MI455X (gfx1250) — matrix operand order repaired and checked by reading, not compiled, not run on hardware
//
#include <hip/hip_runtime.h>

// ---------------------------------------------------------------------------
// Self-attention (B=8, N=2048, D=1024) on gfx1250 via v_wmma_f32_16x16x32_bf16
// ---------------------------------------------------------------------------

typedef __attribute__((ext_vector_type(16))) __bf16 v16bf;
typedef __attribute__((ext_vector_type(8)))  __bf16 v8bf;
typedef __attribute__((ext_vector_type(8)))  float  v8f;
typedef __attribute__((ext_vector_type(4)))  float  v4f;

#define D_MODEL 1024
#define N_TOK   2048
#define N_BATCH 8
#define N_ROWS  (N_BATCH * N_TOK)     // 16384 tokens total

// Scheduling hints: order instruction groups within each k-iteration so all
// loads issue before the WMMA block (forces distinct operand buffers, one
// wait per 8 WMMAs, and overlaps next iteration's loads with WMMA execution).
#if __has_builtin(__builtin_amdgcn_sched_group_barrier)
#define SCHED_GROUP(mask, size, id) __builtin_amdgcn_sched_group_barrier(mask, size, id)
#else
#define SCHED_GROUP(mask, size, id)
#endif
#define SG_VMEM_READ 0x020
#define SG_DS_READ   0x100
#define SG_WMMA      0x008

__device__ __forceinline__ v8f wmma_bf16(v16bf a, v16bf b, v8f c) {
  return __builtin_amdgcn_wmma_f32_16x16x32_bf16(false, a, false, b, (short)0, c,
                                                 false, false);
}

__device__ __forceinline__ v8f vzero8() {
  v8f z;
#pragma unroll
  for (int i = 0; i < 8; ++i) z[i] = 0.0f;
  return z;
}

// A-operand fragment for 16x16x32 bf16 WMMA (A is 16xK row-major in memory).
//  lane<16 : row=lane   , K = [k0+0..15]
//  lane>=16: row=lane-16, K = [k0+16..31]
__device__ __forceinline__ v16bf load_a_frag(const __bf16* base, int ld, int row,
                                             int half, int k0) {
  const __bf16* p = base + (long)row * ld + k0 + half * 16;
  v8bf lo = *(const v8bf*)(p);
  v8bf hi = *(const v8bf*)(p + 8);
  return __builtin_shufflevector(lo, hi, 0, 1, 2, 3, 4, 5, 6, 7,
                                 8, 9, 10, 11, 12, 13, 14, 15);
}

// ---------------------------------------------------------------------------
// Kernel 1: x (f32) -> Xb (bf16), 4 elements per thread
// ---------------------------------------------------------------------------
__global__ __launch_bounds__(256) void cvt_x_kernel(const float* __restrict__ x,
                                                    __bf16* __restrict__ xb) {
  int i = (blockIdx.x * 256 + threadIdx.x) * 4;
  v4f f = *(const v4f*)(x + i);
#pragma unroll
  for (int j = 0; j < 4; ++j) xb[i + j] = (__bf16)f[j];
}

// ---------------------------------------------------------------------------
// Kernel 2: W{q,k,v} (f32, [D][D], x@W convention) -> Wt[3][n][k] = W[k][n] bf16.
// LDS-tiled 64x64 transpose: coalesced global reads AND writes.
// ---------------------------------------------------------------------------
__global__ __launch_bounds__(256) void cvt_w_kernel(const float* __restrict__ wq,
                                                    const float* __restrict__ wk,
                                                    const float* __restrict__ wv,
                                                    __bf16* __restrict__ wt) {
  __shared__ float tile[64][65];   // padded to dodge bank conflicts

  const int blk  = blockIdx.x;
  const int mat  = blk >> 8;              // 256 = 16x16 tiles per matrix
  const int rem  = blk & 255;
  const int tk   = (rem >> 4) * 64;       // source row (k) tile origin
  const int tn   = (rem & 15) * 64;       // source col (n) tile origin

  const float* w = (mat == 0) ? wq : (mat == 1) ? wk : wv;
  __bf16* dst = wt + (long)mat * D_MODEL * D_MODEL;

  const int tx = threadIdx.x & 63;        // column within tile
  const int ty = threadIdx.x >> 6;        // 0..3

#pragma unroll
  for (int r = 0; r < 16; ++r) {
    int k = ty * 16 + r;
    tile[k][tx] = w[(long)(tk + k) * D_MODEL + (tn + tx)];
  }
  __syncthreads();

#pragma unroll
  for (int r = 0; r < 16; ++r) {
    int n = ty * 16 + r;
    dst[(long)(tn + n) * D_MODEL + (tk + tx)] = (__bf16)tile[tx][n];
  }
}

// ---------------------------------------------------------------------------
// Kernel 3: fused QKV projection.
// Each wave: one 16-row x 128-col output strip of one of {Q,K,V}.
// Q,K stored bf16 row-major [row][d]; V stored transposed Vt[b][d][tok] bf16.
// ---------------------------------------------------------------------------
__global__ __launch_bounds__(256) void qkv_kernel(const __bf16* __restrict__ xb,
                                                  const __bf16* __restrict__ wt,
                                                  const float* __restrict__ bq,
                                                  const float* __restrict__ bk,
                                                  const float* __restrict__ bv,
                                                  __bf16* __restrict__ qb,
                                                  __bf16* __restrict__ kb,
                                                  __bf16* __restrict__ vt) {
  const int tid  = threadIdx.x;
  const int lane = tid & 31;
  const int wave = tid >> 5;
  const int col  = lane & 15;
  const int half = lane >> 4;

  long gw = (long)blockIdx.x * 8 + wave;   // [0, 3*1024*8)
  int  mat     = (int)(gw >> 13);          // /8192: 0=Q 1=K 2=V
  int  rem     = (int)(gw & 8191);
  int  rowtile = rem >> 3;                 // 0..1023 : 16-row group
  int  cg      = rem & 7;                  // 0..7    : 128-col group

  const __bf16* A = xb + (long)rowtile * 16 * D_MODEL;
  const __bf16* B = wt + (long)mat * D_MODEL * D_MODEL + (long)cg * 128 * D_MODEL;
  const float* bias = (mat == 0) ? bq : (mat == 1) ? bk : bv;

  v8f acc[8];
#pragma unroll
  for (int t = 0; t < 8; ++t) acc[t] = vzero8();

  for (int k0 = 0; k0 < D_MODEL; k0 += 32) {
    v16bf a = load_a_frag(A, D_MODEL, col, half, k0);
    v16bf bfr[8];
#pragma unroll
    for (int t = 0; t < 8; ++t)
      bfr[t] = *(const v16bf*)(B + (long)(t * 16 + col) * D_MODEL + k0 + half * 16);
#pragma unroll
    for (int t = 0; t < 8; ++t) acc[t] = wmma_bf16(a, bfr[t], acc[t]);
    SCHED_GROUP(SG_VMEM_READ, 18, 0);   // A (2) + B (16) loads first...
    SCHED_GROUP(SG_WMMA,       8, 0);   // ...then the 8 WMMAs back-to-back
  }

#pragma unroll
  for (int t = 0; t < 8; ++t) {
    int   n  = cg * 128 + t * 16 + col;
    float bs = bias[n];
    if (mat < 2) {
      __bf16* dst = ((mat == 0) ? qb : kb) + (long)rowtile * 16 * D_MODEL + n;
#pragma unroll
      for (int r = 0; r < 8; ++r) {
        int row = half * 8 + r;
        dst[(long)row * D_MODEL] = (__bf16)(acc[t][r] + bs);
      }
    } else {
      long rowg = (long)rowtile * 16;
      int  bb   = (int)(rowg >> 11);                // batch
      int  t0   = (int)(rowg & 2047) + half * 8;    // token base (contiguous 8)
      v8bf pack;
#pragma unroll
      for (int r = 0; r < 8; ++r) pack[r] = (__bf16)(acc[t][r] + bs);
      *(v8bf*)(vt + ((long)bb << 21) + (long)n * N_TOK + t0) = pack;
    }
  }
}

// ---------------------------------------------------------------------------
// Kernel 4: attention.  One block (8 waves / 256 threads) per (batch, 16-query
// tile).  Phase1: S = Q K^T / 32 -> LDS (bf16).  Phase2: softmax in place.
// Phase3: out = P @ V using Vt, f32 direct to d_out.
// ---------------------------------------------------------------------------
__global__ __launch_bounds__(256) void attn_kernel(const __bf16* __restrict__ qb,
                                                   const __bf16* __restrict__ kb,
                                                   const __bf16* __restrict__ vt,
                                                   float* __restrict__ out) {
  __shared__ __attribute__((aligned(32))) __bf16 SP[16 * N_TOK];  // 64 KB

  const int tid  = threadIdx.x;
  const int lane = tid & 31;
  const int wave = tid >> 5;
  const int col  = lane & 15;
  const int half = lane >> 4;

  const int  b  = blockIdx.x >> 7;    // batch
  const int  qt = blockIdx.x & 127;   // query tile
  const long q0 = (long)b * N_TOK + qt * 16;

  const __bf16* Qtile = qb + q0 * D_MODEL;
  const __bf16* Kbase = kb + (long)b * N_TOK * D_MODEL;
  const __bf16* Vbase = vt + ((long)b << 21);   // [d][tok]

  // ---- Phase 1: scores. Wave w covers keys [w*256, w*256+256), as two
  // groups of 8 key tiles (keeps registers in check for occupancy).
#pragma unroll 1
  for (int g = 0; g < 2; ++g) {
    const int kbase0 = wave * 256 + g * 128;
    v8f acc[8];
#pragma unroll
    for (int t = 0; t < 8; ++t) acc[t] = vzero8();

    for (int k0 = 0; k0 < D_MODEL; k0 += 32) {
      v16bf a = load_a_frag(Qtile, D_MODEL, col, half, k0);
      v16bf bfr[8];
#pragma unroll
      for (int t = 0; t < 8; ++t) {
        int key = kbase0 + t * 16 + col;
        bfr[t]  = *(const v16bf*)(Kbase + (long)key * D_MODEL + k0 + half * 16);
      }
#pragma unroll
      for (int t = 0; t < 8; ++t) acc[t] = wmma_bf16(a, bfr[t], acc[t]);
      SCHED_GROUP(SG_VMEM_READ, 18, 0);
      SCHED_GROUP(SG_WMMA,       8, 0);
    }
#pragma unroll
    for (int t = 0; t < 8; ++t) {
      int key = kbase0 + t * 16 + col;
#pragma unroll
      for (int r = 0; r < 8; ++r) {
        int row = half * 8 + r;
        SP[row * N_TOK + key] = (__bf16)(acc[t][r] * 0.03125f);  // 1/sqrt(1024)
      }
    }
  }
  __syncthreads();

  // ---- Phase 2: softmax per query row (16 threads per row, 128 elems each).
  {
    int row = tid >> 4;
    int j   = tid & 15;
    __bf16* srow = SP + row * N_TOK + j * 128;

    float m = -3.0e38f;
    for (int i = 0; i < 128; ++i) m = fmaxf(m, (float)srow[i]);
#pragma unroll
    for (int o = 8; o > 0; o >>= 1) m = fmaxf(m, __shfl_xor(m, o, 32));

    float s = 0.0f;
    for (int i = 0; i < 128; ++i) s += __expf((float)srow[i] - m);
#pragma unroll
    for (int o = 8; o > 0; o >>= 1) s += __shfl_xor(s, o, 32);

    float inv = 1.0f / s;
    for (int i = 0; i < 128; ++i)
      srow[i] = (__bf16)(__expf((float)srow[i] - m) * inv);
  }
  __syncthreads();

  // ---- Phase 3: out = P @ V.  Wave w covers d-slice [w*128, w*128+128).
  {
    v8f acc[8];
#pragma unroll
    for (int t = 0; t < 8; ++t) acc[t] = vzero8();

    const int d0 = wave * 128;
    for (int m0 = 0; m0 < N_TOK; m0 += 32) {
      v16bf a = load_a_frag(SP, N_TOK, col, half, m0);   // P fragment from LDS
      v16bf bfr[8];
#pragma unroll
      for (int t = 0; t < 8; ++t) {
        int d  = d0 + t * 16 + col;
        bfr[t] = *(const v16bf*)(Vbase + (long)d * N_TOK + m0 + half * 16);
      }
#pragma unroll
      for (int t = 0; t < 8; ++t) acc[t] = wmma_bf16(a, bfr[t], acc[t]);
      SCHED_GROUP(SG_DS_READ,    2, 0);   // A fragment from LDS
      SCHED_GROUP(SG_VMEM_READ, 16, 0);   // 8 x 32B V fragments
      SCHED_GROUP(SG_WMMA,       8, 0);
    }
#pragma unroll
    for (int t = 0; t < 8; ++t) {
      int d = d0 + t * 16 + col;
#pragma unroll
      for (int r = 0; r < 8; ++r) {
        int row = half * 8 + r;
        out[(q0 + row) * D_MODEL + d] = acc[t][r];
      }
    }
  }
}

// ---------------------------------------------------------------------------
extern "C" void kernel_launch(void* const* d_in, const int* in_sizes, int n_in,
                              void* d_out, int out_size, void* d_ws, size_t ws_size,
                              hipStream_t stream) {
  (void)in_sizes; (void)n_in; (void)out_size; (void)ws_size;

  const float* x  = (const float*)d_in[0];
  const float* Wq = (const float*)d_in[1];
  const float* bq = (const float*)d_in[2];
  const float* Wk = (const float*)d_in[3];
  const float* bk = (const float*)d_in[4];
  const float* Wv = (const float*)d_in[5];
  const float* bv = (const float*)d_in[6];
  float* out = (float*)d_out;

  // Workspace carve-up (bf16 everywhere): ~134 MB total.
  char* ws = (char*)d_ws;
  __bf16* Xb = (__bf16*)ws; ws += (size_t)N_ROWS * D_MODEL * 2;           // 32 MB
  __bf16* Wt = (__bf16*)ws; ws += (size_t)3 * D_MODEL * D_MODEL * 2;      //  6 MB
  __bf16* Qb = (__bf16*)ws; ws += (size_t)N_ROWS * D_MODEL * 2;           // 32 MB
  __bf16* Kb = (__bf16*)ws; ws += (size_t)N_ROWS * D_MODEL * 2;           // 32 MB
  __bf16* Vt = (__bf16*)ws;                                               // 32 MB

  cvt_x_kernel<<<(N_ROWS * D_MODEL) / (256 * 4), 256, 0, stream>>>(x, Xb);
  cvt_w_kernel<<<3 * 16 * 16, 256, 0, stream>>>(Wq, Wk, Wv, Wt);
  qkv_kernel<<<3072, 256, 0, stream>>>(Xb, Wt, bq, bk, bv, Qb, Kb, Vt);
  attn_kernel<<<N_BATCH * 128, 256, 0, stream>>>(Qb, Kb, Vt, out);
}
